// RNNAttentionDecoder_29214367547953
// MI455X (gfx1250) — compile-verified, matrix operand order checked by reading, not run on hardware
//
#include <hip/hip_runtime.h>

// ---------------- problem constants ----------------
static constexpr int kT = 50;
static constexpr int kB = 64;
static constexpr int kS = 50;
static constexpr int kE = 512;
static constexpr int kH = 512;
static constexpr int kV = 32000;

// ---------------- WMMA types ----------------
typedef __attribute__((ext_vector_type(16))) __bf16 v16bf;
typedef __attribute__((ext_vector_type(8)))  float  v8f;

union FragB16 { v16bf v; uint4 q[2]; };

__device__ __forceinline__ unsigned short f2b(float f) {
  unsigned u = __float_as_uint(f);
  unsigned r = u + 0x7FFFu + ((u >> 16) & 1u);   // round-to-nearest-even
  return (unsigned short)(r >> 16);
}
__device__ __forceinline__ float b2f(unsigned short s) {
  return __uint_as_float(((unsigned)s) << 16);
}
__device__ __forceinline__ float sigm(float x) { return 1.f / (1.f + __expf(-x)); }

// Load a 16x32 bf16 fragment from a row-major [rows][stride] bf16 buffer.
// Per CDNA5 ISA 16-bit A layout: lane = half*16 + m holds row m,
// k = {k0 + half*8 .. +7} and {k0+16+half*8 .. +7}  (two 16B chunks).
// The same pattern serves B when B is stored as [N][K] (i.e. W row-major).
__device__ __forceinline__ v16bf ldfrag(const unsigned short* __restrict__ base,
                                        int stride, int row0, int k0, int lane) {
  const unsigned short* p =
      base + (size_t)(row0 + (lane & 15)) * stride + k0 + ((lane >> 4) << 3);
  FragB16 f;
  f.q[0] = *reinterpret_cast<const uint4*>(p);
  f.q[1] = *reinterpret_cast<const uint4*>(p + 16);
  return f.v;
}

__device__ __forceinline__ v8f wmma_bf16(v16bf a, v16bf b, v8f c) {
  return __builtin_amdgcn_wmma_f32_16x16x32_bf16(false, a, false, b, (short)0, c,
                                                 false, false);
}

// ---------------- helper kernels ----------------
__global__ void k_cvt(const float* __restrict__ src, unsigned short* __restrict__ dst,
                      int n) {
  int i = blockIdx.x * blockDim.x + threadIdx.x;
  if (i < n) dst[i] = f2b(src[i]);
}

__global__ void k_embed(const int* __restrict__ idw, const int* __restrict__ ids,
                        const float* __restrict__ wemb, const float* __restrict__ semb,
                        unsigned short* __restrict__ dst) {
  int i = blockIdx.x * blockDim.x + threadIdx.x;
  if (i >= kT * kB * kE) return;
  int e  = i % kE;
  int tb = i / kE;  // t*kB + b
  int w = idw[tb], s = ids[tb];
  dst[i] = f2b(wemb[(size_t)w * kE + e] + semb[(size_t)s * kE + e]);
}

// ---------------- persistent recurrent decoder ----------------
// One workgroup, 512 threads = 16 wave32s. LDS (dynamic, 275456 B):
//   h0, h1, qwc, outp : [64][512] bf16 (64 KB each)   att : [64][52] f32
__global__ void __launch_bounds__(512) k_decoder(
    const unsigned short* __restrict__ emb,   // [T][B][E] bf16
    const unsigned short* __restrict__ Wih,   // [3H][E+H] bf16
    const unsigned short* __restrict__ Whh,   // [3H][H]  bf16
    const unsigned short* __restrict__ Wal,   // [H][H]
    const unsigned short* __restrict__ Wco,   // [H][H]
    const unsigned short* __restrict__ Wqu,   // [H][H]
    unsigned short* __restrict__ outall,      // [T+1][B][H] bf16 (rows 1..T written)
    const float* __restrict__ hidden0,        // [B][H]
    const float* __restrict__ prev_out,       // [B][H]
    const float* __restrict__ context,        // [S][B][H] f32
    const unsigned char* __restrict__ cmask,  // [B][S] bool
    const float* __restrict__ b_ih, const float* __restrict__ b_hh,
    float* __restrict__ d_out) {
  extern __shared__ unsigned char smem[];
  unsigned short* h0   = (unsigned short*)smem;
  unsigned short* h1   = h0 + kB * kH;
  unsigned short* qwc  = h1 + kB * kH;
  unsigned short* outp = qwc + kB * kH;
  float* att = (float*)(outp + kB * kH);  // stride 52

  const int tid = threadIdx.x;
  const int lane = tid & 31, wave = tid >> 5;

  for (int i = tid; i < kB * kH; i += 512) {
    h0[i]   = f2b(hidden0[i]);
    outp[i] = f2b(prev_out[i]);
  }
  __syncthreads();

  unsigned short* hc = h0;
  unsigned short* hn = h1;

  const size_t hid_off  = (size_t)kT * kB * kV;
  const size_t outf_off = hid_off + (size_t)kB * kH;

  for (int t = 0; t < kT; ++t) {
    const unsigned short* embt = emb + (size_t)t * kB * kE;

    // -------- GRU gates: gi = x Wih^T, gh = h Whh^T, combine --------
    // wave handles column blocks jb = wave, wave+16; all 4 m-blocks per job
    for (int jj = 0; jj < 2; ++jj) {
      int jb = wave + jj * 16;
      int n0 = jb * 16;
      v8f aR[4], aZ[4], aNi[4], aNh[4];
#pragma unroll
      for (int mb = 0; mb < 4; ++mb) {
        v8f z = {};
        aR[mb] = z; aZ[mb] = z; aNi[mb] = z; aNh[mb] = z;
      }
      // x = emb_t portion (global A-fragments), k in [0, kE)
      for (int k0 = 0; k0 < kE; k0 += 32) {
        v16bf Br = ldfrag(Wih, kE + kH, 0 * kH + n0, k0, lane);
        v16bf Bz = ldfrag(Wih, kE + kH, 1 * kH + n0, k0, lane);
        v16bf Bn = ldfrag(Wih, kE + kH, 2 * kH + n0, k0, lane);
#pragma unroll
        for (int mb = 0; mb < 4; ++mb) {
          v16bf A = ldfrag(embt, kE, mb * 16, k0, lane);
          aR[mb]  = wmma_bf16(A, Br, aR[mb]);
          aZ[mb]  = wmma_bf16(A, Bz, aZ[mb]);
          aNi[mb] = wmma_bf16(A, Bn, aNi[mb]);
        }
      }
      // x = prev-output portion (LDS A-fragments), k in [kE, kE+kH)
      for (int k0 = 0; k0 < kH; k0 += 32) {
        v16bf Br = ldfrag(Wih, kE + kH, 0 * kH + n0, kE + k0, lane);
        v16bf Bz = ldfrag(Wih, kE + kH, 1 * kH + n0, kE + k0, lane);
        v16bf Bn = ldfrag(Wih, kE + kH, 2 * kH + n0, kE + k0, lane);
#pragma unroll
        for (int mb = 0; mb < 4; ++mb) {
          v16bf A = ldfrag(outp, kH, mb * 16, k0, lane);
          aR[mb]  = wmma_bf16(A, Br, aR[mb]);
          aZ[mb]  = wmma_bf16(A, Bz, aZ[mb]);
          aNi[mb] = wmma_bf16(A, Bn, aNi[mb]);
        }
      }
      // gh = h @ Whh^T (LDS A-fragments)
      for (int k0 = 0; k0 < kH; k0 += 32) {
        v16bf Br = ldfrag(Whh, kH, 0 * kH + n0, k0, lane);
        v16bf Bz = ldfrag(Whh, kH, 1 * kH + n0, k0, lane);
        v16bf Bn = ldfrag(Whh, kH, 2 * kH + n0, k0, lane);
#pragma unroll
        for (int mb = 0; mb < 4; ++mb) {
          v16bf A = ldfrag(hc, kH, mb * 16, k0, lane);
          aR[mb]  = wmma_bf16(A, Br, aR[mb]);
          aZ[mb]  = wmma_bf16(A, Bz, aZ[mb]);
          aNh[mb] = wmma_bf16(A, Bn, aNh[mb]);
        }
      }
      int col = n0 + (lane & 15);
      float bir = b_ih[col],          bhr = b_hh[col];
      float biz = b_ih[kH + col],     bhz = b_hh[kH + col];
      float bin = b_ih[2 * kH + col], bhn = b_hh[2 * kH + col];
#pragma unroll
      for (int mb = 0; mb < 4; ++mb) {
        int mbase = mb * 16 + ((lane >> 4) << 3);
#pragma unroll
        for (int r = 0; r < 8; ++r) {
          int m = mbase + r;
          float ho = b2f(hc[m * kH + col]);
          float rg = sigm(aR[mb][r] + bir + bhr);
          float zg = sigm(aZ[mb][r] + biz + bhz);
          float ng = tanhf(aNi[mb][r] + bin + rg * (aNh[mb][r] + bhn));
          float hv = (1.f - zg) * ng + zg * ho;
          hn[m * kH + col] = f2b(hv);
          if (t == kT - 1) d_out[hid_off + (size_t)m * kH + col] = hv;
        }
      }
    }
    __syncthreads();

    // -------- q = h_new @ W_align^T -> qwc --------
    for (int jj = 0; jj < 2; ++jj) {
      int n0 = (wave + jj * 16) * 16;
      v8f acc[4];
#pragma unroll
      for (int mb = 0; mb < 4; ++mb) { v8f z = {}; acc[mb] = z; }
      for (int k0 = 0; k0 < kH; k0 += 32) {
        v16bf Bm = ldfrag(Wal, kH, n0, k0, lane);
#pragma unroll
        for (int mb = 0; mb < 4; ++mb) {
          v16bf A = ldfrag(hn, kH, mb * 16, k0, lane);
          acc[mb] = wmma_bf16(A, Bm, acc[mb]);
        }
      }
      int col = n0 + (lane & 15);
#pragma unroll
      for (int mb = 0; mb < 4; ++mb) {
        int mbase = mb * 16 + ((lane >> 4) << 3);
#pragma unroll
        for (int r = 0; r < 8; ++r) qwc[(mbase + r) * kH + col] = f2b(acc[mb][r]);
      }
    }
    __syncthreads();

    // -------- attention scores: att[b][s] = context[s,b,:] . q[b,:] --------
    for (int p = tid; p < kB * kS; p += 512) {
      int b = p / kS, s = p % kS;
      const float* cp = context + ((size_t)s * kB + b) * kH;
      float acc = 0.f;
      for (int h = 0; h < kH; ++h) acc += b2f(qwc[b * kH + h]) * cp[h];
      att[b * 52 + s] = cmask[b * kS + s] ? -1e9f : acc;
    }
    __syncthreads();

    // -------- softmax over S per row --------
    if (tid < kB) {
      float mx = -1e30f;
      for (int s = 0; s < kS; ++s) mx = fmaxf(mx, att[tid * 52 + s]);
      float sm = 0.f;
      for (int s = 0; s < kS; ++s) {
        float e = __expf(att[tid * 52 + s] - mx);
        att[tid * 52 + s] = e;
        sm += e;
      }
      float inv = 1.f / sm;
      for (int s = 0; s < kS; ++s) att[tid * 52 + s] *= inv;
    }
    __syncthreads();

    // -------- weighted context -> qwc (overwrites q) --------
    for (int i = tid; i < kB * kH; i += 512) {
      int b = i >> 9, h = i & (kH - 1);
      float acc = 0.f;
      for (int s = 0; s < kS; ++s)
        acc += att[b * 52 + s] * context[((size_t)s * kB + b) * kH + h];
      qwc[i] = f2b(acc);
    }
    __syncthreads();

    // -------- out = tanh(wc Wco^T + h Wqu^T) --------
    for (int jj = 0; jj < 2; ++jj) {
      int n0 = (wave + jj * 16) * 16;
      v8f acc[4];
#pragma unroll
      for (int mb = 0; mb < 4; ++mb) { v8f z = {}; acc[mb] = z; }
      for (int k0 = 0; k0 < kH; k0 += 32) {
        v16bf Bc = ldfrag(Wco, kH, n0, k0, lane);
        v16bf Bq = ldfrag(Wqu, kH, n0, k0, lane);
#pragma unroll
        for (int mb = 0; mb < 4; ++mb) {
          v16bf Aw = ldfrag(qwc, kH, mb * 16, k0, lane);
          v16bf Ah = ldfrag(hn, kH, mb * 16, k0, lane);
          acc[mb] = wmma_bf16(Aw, Bc, acc[mb]);
          acc[mb] = wmma_bf16(Ah, Bq, acc[mb]);
        }
      }
      int col = n0 + (lane & 15);
#pragma unroll
      for (int mb = 0; mb < 4; ++mb) {
        int mbase = mb * 16 + ((lane >> 4) << 3);
#pragma unroll
        for (int r = 0; r < 8; ++r) {
          int m = mbase + r;
          float v = tanhf(acc[mb][r]);
          unsigned short bv = f2b(v);
          outp[m * kH + col] = bv;
          outall[((size_t)(t + 1) * kB + m) * kH + col] = bv;
          if (t == kT - 1) d_out[outf_off + (size_t)m * kH + col] = v;
        }
      }
    }
    // swap hidden buffers
    { unsigned short* tp = hc; hc = hn; hn = tp; }
    __syncthreads();
  }
}

// ---------------- generator: scores = log_softmax(out @ Wgen^T + b) ----------------
// One block per 16 score rows; 16 waves split the 2000 column tiles.
__global__ void __launch_bounds__(512) k_gen(const unsigned short* __restrict__ outall,
                                             const unsigned short* __restrict__ Wgen,
                                             const float* __restrict__ b_gen,
                                             float* __restrict__ d_out) {
  __shared__ unsigned short Ablk[16 * kH];  // 16 KB
  __shared__ float statsM[16][16];
  __shared__ float statsS[16][16];
  __shared__ float denom[16];

  const int tid = threadIdx.x, lane = tid & 31, wave = tid >> 5;
  const int r0 = blockIdx.x * 16;

  // Async-stage the 16x512 bf16 A block into LDS (rows r0.. map to outall rows 64+r0..)
  {
    const unsigned char* g =
        (const unsigned char*)(outall + (size_t)(kB + r0) * kH) + tid * 16;
    unsigned lds = (unsigned)(unsigned long long)(uintptr_t)&Ablk[0] + (unsigned)(tid * 16);
    asm volatile("global_load_async_to_lds_b128 %0, %1, off" ::"v"(lds), "v"(g)
                 : "memory");
    asm volatile("global_load_async_to_lds_b128 %0, %1, off" ::"v"(lds + 8192u),
                 "v"(g + 8192)
                 : "memory");
    asm volatile("s_wait_asynccnt 0" ::: "memory");
  }
  __syncthreads();

  float mx[8], sm[8];
#pragma unroll
  for (int r = 0; r < 8; ++r) { mx[r] = -1e30f; sm[r] = 0.f; }

  float* outBase = d_out + (size_t)r0 * kV;

  for (int j = wave; j < kV / 16; j += 16) {
    int n0 = j * 16;
    if (j + 16 < kV / 16)
      __builtin_prefetch(Wgen + (size_t)(n0 + 256) * kH, 0, 0);
    v8f acc = {};
    for (int k0 = 0; k0 < kH; k0 += 32) {
      v16bf A  = ldfrag(Ablk, kH, 0, k0, lane);
      v16bf Bm = ldfrag(Wgen, kH, n0, k0, lane);
      acc = wmma_bf16(A, Bm, acc);
    }
    int n = n0 + (lane & 15);
    int mb = (lane >> 4) << 3;
    float bias = b_gen[n];
#pragma unroll
    for (int r = 0; r < 8; ++r) {
      float v = acc[r] + bias;
      outBase[(size_t)(mb + r) * kV + n] = v;
      float nm = fmaxf(mx[r], v);
      sm[r] = sm[r] * __expf(mx[r] - nm) + __expf(v - nm);
      mx[r] = nm;
    }
  }
  // reduce (max,sumexp) across the 16-lane half that shares each row
#pragma unroll
  for (int r = 0; r < 8; ++r) {
#pragma unroll
    for (int msk = 1; msk < 16; msk <<= 1) {
      float om = __shfl_xor(mx[r], msk, 32);
      float os = __shfl_xor(sm[r], msk, 32);
      float nm = fmaxf(mx[r], om);
      sm[r] = sm[r] * __expf(mx[r] - nm) + os * __expf(om - nm);
      mx[r] = nm;
    }
  }
  if ((lane & 15) == 0) {
    int base = (lane >> 4) << 3;
#pragma unroll
    for (int r = 0; r < 8; ++r) {
      statsM[wave][base + r] = mx[r];
      statsS[wave][base + r] = sm[r];
    }
  }
  __threadfence();  // make pass-1 logit stores visible to the normalize sweep
  __syncthreads();
  if (tid < 16) {
    float M = -1e30f, Ss = 0.f;
    for (int w = 0; w < 16; ++w) {
      float om = statsM[w][tid], os = statsS[w][tid];
      float nm = fmaxf(M, om);
      Ss = Ss * __expf(M - nm) + os * __expf(om - nm);
      M = nm;
    }
    denom[tid] = M + __logf(Ss);
  }
  __syncthreads();
  for (int m = 0; m < 16; ++m) {
    float dn = denom[m];
    float* row = outBase + (size_t)m * kV;
    for (int i = tid; i < kV; i += 512) row[i] -= dn;
  }
}

// ---------------- host launcher ----------------
extern "C" void kernel_launch(void* const* d_in, const int* in_sizes, int n_in,
                              void* d_out, int out_size, void* d_ws, size_t ws_size,
                              hipStream_t stream) {
  (void)in_sizes; (void)n_in; (void)out_size; (void)ws_size;
  const int*   ids_word    = (const int*)d_in[0];
  const int*   ids_special = (const int*)d_in[1];
  const unsigned char* cmask = (const unsigned char*)d_in[2];
  const float* hidden      = (const float*)d_in[3];
  const float* context     = (const float*)d_in[4];
  const float* prev_output = (const float*)d_in[5];
  const float* word_emb    = (const float*)d_in[6];
  const float* special_emb = (const float*)d_in[7];
  const float* W_ih        = (const float*)d_in[8];
  const float* b_ih        = (const float*)d_in[9];
  const float* W_hh        = (const float*)d_in[10];
  const float* b_hh        = (const float*)d_in[11];
  const float* W_align     = (const float*)d_in[12];
  const float* W_context   = (const float*)d_in[13];
  const float* W_query     = (const float*)d_in[14];
  const float* W_gen       = (const float*)d_in[15];
  const float* b_gen       = (const float*)d_in[16];
  float* out = (float*)d_out;

  // workspace carve (bf16 buffers), all 16B aligned
  unsigned short* ws = (unsigned short*)d_ws;
  size_t off = 0;
  unsigned short* emb    = ws + off; off += (size_t)kT * kB * kE;        // 1.6M
  unsigned short* Wih    = ws + off; off += (size_t)3 * kH * (kE + kH);  // 1.57M
  unsigned short* Whh    = ws + off; off += (size_t)3 * kH * kH;
  unsigned short* Wal    = ws + off; off += (size_t)kH * kH;
  unsigned short* Wco    = ws + off; off += (size_t)kH * kH;
  unsigned short* Wqu    = ws + off; off += (size_t)kH * kH;
  unsigned short* Wgen   = ws + off; off += (size_t)kV * kH;             // 16.4M
  unsigned short* outall = ws + off; off += (size_t)(kT + 1) * kB * kH;

  k_embed<<<(kT * kB * kE + 255) / 256, 256, 0, stream>>>(ids_word, ids_special,
                                                          word_emb, special_emb, emb);
  auto cvt = [&](const float* s, unsigned short* dd, int n) {
    k_cvt<<<(n + 255) / 256, 256, 0, stream>>>(s, dd, n);
  };
  cvt(W_ih, Wih, 3 * kH * (kE + kH));
  cvt(W_hh, Whh, 3 * kH * kH);
  cvt(W_align, Wal, kH * kH);
  cvt(W_context, Wco, kH * kH);
  cvt(W_query, Wqu, kH * kH);
  cvt(W_gen, Wgen, kV * kH);

  size_t smem = (size_t)4 * kB * kH * sizeof(unsigned short) + (size_t)kB * 52 * sizeof(float);
  k_decoder<<<1, 512, smem, stream>>>(emb, Wih, Whh, Wal, Wco, Wqu, outall, hidden,
                                      prev_output, context, cmask, b_ih, b_hh, out);
  k_gen<<<(kT * kB) / 16, 512, 0, stream>>>(outall, Wgen, b_gen, out);
}
